// SABlock_3994319585683
// MI455X (gfx1250) — matrix operand order repaired and checked by reading, not compiled, not run on hardware
//
#include <hip/hip_runtime.h>
#include <hip/hip_bf16.h>

// ---------------------------------------------------------------------------
// MI455X (gfx1250) self-attention block.
// Compute-bound (~69 GFLOP vs ~50MB traffic @ 23.3TB/s) -> all GEMMs through
// v_wmma_f32_16x16x32_bf16 (f32 accum); LN/softmax in f32 VALU.
// GEMM double-buffers 128x128 block tiles in LDS using gfx1250 async
// global->LDS copies: next tile's copies are issued before computing the
// current one, and "s_wait_asynccnt 4" (in-order completion) lets the new
// copies stay in flight while guaranteeing the current tile has landed.
// ---------------------------------------------------------------------------

typedef __attribute__((ext_vector_type(16))) __bf16 bf16x16;
typedef __attribute__((ext_vector_type(8)))  float  f32x8;

union Frag16 { bf16x16 v; uint4 q[2]; };

__device__ inline f32x8 wmma_bf16(bf16x16 a, bf16x16 b, f32x8 c) {
    return __builtin_amdgcn_wmma_f32_16x16x32_bf16(
        /*neg_a=*/false, a, /*neg_b=*/false, b,
        /*c_mod=*/(short)0, c, /*reuse_a=*/false, /*reuse_b=*/false);
}

// --- async global -> LDS copy (16B per lane), with safe fallback -----------
#if __has_builtin(__builtin_amdgcn_global_load_async_to_lds_b128)
#define HAS_ASYNC_LDS 1
typedef int v4i_ __attribute__((vector_size(16)));
typedef __attribute__((address_space(1))) v4i_ v4i_gbl;   // global AS pointee
typedef __attribute__((address_space(3))) v4i_ v4i_lds;   // LDS AS pointee
#else
#define HAS_ASYNC_LDS 0
#endif

__device__ inline void async_copy_b128(const void* gsrc, void* lds) {
#if HAS_ASYNC_LDS
    // generic->as(1): identical bit pattern; generic->as(3): low 32 bits are
    // the LDS offset (ISA 10.2 aperture mapping).
    v4i_gbl* g = (v4i_gbl*)(unsigned long long)gsrc;
    v4i_lds* l = (v4i_lds*)(unsigned int)(unsigned long long)lds;
    __builtin_amdgcn_global_load_async_to_lds_b128(g, l, 0, 0);
#else
    *(uint4*)lds = *(const uint4*)gsrc;
#endif
}

// Wait until ALL of this wave's async copies have landed.
__device__ inline void async_wait_all() {
#if HAS_ASYNC_LDS
    asm volatile("s_wait_asynccnt 0" ::: "memory");
#endif
}

// Wait until all but the 4 most recent async copies have landed (async loads
// complete in order, so with 8 outstanding this guarantees the oldest 4).
__device__ inline void async_wait_except4() {
#if HAS_ASYNC_LDS
    asm volatile("s_wait_asynccnt 4" ::: "memory");
#endif
}

// A-fragment (16x32 bf16, row-major, leading dim ld in elements):
//  lanes 0-15: row=lane, K 0..7 / 16..23 ; lanes 16-31: row=lane-16, K 8..15 / 24..31
__device__ inline bf16x16 load_a_frag(const __bf16* base, int ld) {
    int lane = threadIdx.x & 31;
    int row  = lane & 15;
    int half = lane >> 4;
    const __bf16* p = base + (size_t)row * ld + half * 8;
    Frag16 f;
    f.q[0] = *(const uint4*)(p);
    f.q[1] = *(const uint4*)(p + 16);
    return f.v;
}

// B-fragment (32x16 bf16) from column-contiguous source: Bt[col][k], ld = K.
//  lanes 0-15: col=lane, K 0..7 / 16..23 ; lanes 16-31: col=lane-16, K 8..15 / 24..31
__device__ inline bf16x16 load_b_frag(const __bf16* base, int ld) {
    int lane = threadIdx.x & 31;
    int col  = lane & 15;
    int half = lane >> 4;
    const __bf16* p = base + (size_t)col * ld + half * 8;
    Frag16 f;
    f.q[0] = *(const uint4*)(p);
    f.q[1] = *(const uint4*)(p + 16);
    return f.v;
}

// ---------------------------------------------------------------------------
// f32 -> bf16 conversion
// ---------------------------------------------------------------------------
__global__ void f32_to_bf16_kernel(const float* __restrict__ in,
                                   __bf16* __restrict__ out, size_t n) {
    size_t i = ((size_t)blockIdx.x * blockDim.x + threadIdx.x) * 4;
    if (i + 3 < n) {
        float4 v = *(const float4*)(in + i);
        out[i + 0] = (__bf16)v.x;
        out[i + 1] = (__bf16)v.y;
        out[i + 2] = (__bf16)v.z;
        out[i + 3] = (__bf16)v.w;
    }
}

// ---------------------------------------------------------------------------
// GEMM: D[M,N] = A[M,K] * Bt[N,K]^T (+ bias[N]); bf16 in, f32 out.
// Block 256 thr = 8 waves (4x2); block tile 128x128, wave tile 32x64, BK=32.
// Double-buffered LDS staging via async copies; rows padded +8 bf16 (16B)
// to break the 64B power-of-two bank stride while keeping 16B alignment.
// ---------------------------------------------------------------------------
#define BK   32
#define LDP  40   // padded row length in elements (80B, 16B-aligned)

__global__ void __launch_bounds__(256)
gemm_bf16_wmma(const __bf16* __restrict__ A, const __bf16* __restrict__ Bt,
               const float* __restrict__ bias, float* __restrict__ Dst,
               int M, int N, int K) {
    __shared__ __bf16 Asm[2][128][LDP];
    __shared__ __bf16 Bsm[2][128][LDP];

    int tid  = threadIdx.x;
    int wave = tid >> 5;
    int lane = tid & 31;
    int wm   = wave & 3;           // 0..3 -> row group
    int wn   = wave >> 2;          // 0..1 -> col group
    int m0   = blockIdx.x * 128;
    int n0   = blockIdx.y * 128;
    int wm0  = wm * 32;
    int wn0  = wn * 64;

    // Tile copy mapping: 128 rows x 32 elems = 512 chunks of 8 bf16 (16B),
    // 2 chunks per thread -> 4 async copies per thread per tile.
    int c0 = tid * 2, c1 = tid * 2 + 1;
    int rA0 = c0 >> 2, sA0 = (c0 & 3) * 8;
    int rA1 = c1 >> 2, sA1 = (c1 & 3) * 8;

    auto issue_tile = [&](int k0, int buf) {
        async_copy_b128(A  + (size_t)(m0 + rA0) * K + k0 + sA0, &Asm[buf][rA0][sA0]);
        async_copy_b128(A  + (size_t)(m0 + rA1) * K + k0 + sA1, &Asm[buf][rA1][sA1]);
        async_copy_b128(Bt + (size_t)(n0 + rA0) * K + k0 + sA0, &Bsm[buf][rA0][sA0]);
        async_copy_b128(Bt + (size_t)(n0 + rA1) * K + k0 + sA1, &Bsm[buf][rA1][sA1]);
    };

    f32x8 acc[2][4] = {};

    issue_tile(0, 0);                        // prime buffer 0
    const int nk = K / BK;
    for (int kt = 0; kt < nk; kt++) {
        int buf = kt & 1;
        if (kt + 1 < nk) {
            // Overwrite-safety: buf^1 was last read in iteration kt-1, whose
            // closing barrier ordered all readers before we got here.
            issue_tile((kt + 1) * BK, buf ^ 1);
            async_wait_except4();            // current tile landed; next in flight
        } else {
            async_wait_all();
        }
        __syncthreads();

        bf16x16 af[2], bfr[4];
#pragma unroll
        for (int i = 0; i < 2; i++)
            af[i] = load_a_frag(&Asm[buf][wm0 + i * 16][0], LDP);
#pragma unroll
        for (int j = 0; j < 4; j++)
            bfr[j] = load_b_frag(&Bsm[buf][wn0 + j * 16][0], LDP);
#pragma unroll
        for (int i = 0; i < 2; i++)
#pragma unroll
            for (int j = 0; j < 4; j++)
                acc[i][j] = wmma_bf16(af[i], bfr[j], acc[i][j]);

        __syncthreads();
    }

    int col = lane & 15, half = lane >> 4;
#pragma unroll
    for (int i = 0; i < 2; i++)
#pragma unroll
        for (int j = 0; j < 4; j++) {
            int n = n0 + wn0 + j * 16 + col;
            float bv = bias ? bias[n] : 0.0f;
#pragma unroll
            for (int v = 0; v < 8; v++) {
                int m = m0 + wm0 + i * 16 + v + 8 * half;
                Dst[(size_t)m * N + n] = acc[i][j][v] + bv;
            }
        }
}

// ---------------------------------------------------------------------------
// Fused q/k LayerNorm over C per token + scatter q,k,v to bf16 [B,H,N,D].
// ---------------------------------------------------------------------------
__global__ void __launch_bounds__(256)
qk_ln_kernel(const float* __restrict__ qkv,
             const float* __restrict__ qg, const float* __restrict__ qbeta,
             const float* __restrict__ kg, const float* __restrict__ kbeta,
             __bf16* __restrict__ qout, __bf16* __restrict__ kout,
             __bf16* __restrict__ vout,
             int B, int N, int C, int H, int D) {
    int t = blockIdx.x;
    int b = t / N, n = t % N;
    const float* row = qkv + (size_t)t * 3 * C;
    int tid = threadIdx.x;

    float sq = 0.f, sqq = 0.f, sk = 0.f, skk = 0.f;
    for (int c = tid; c < C; c += 256) {
        float qv = row[c];
        float kv = row[C + c];
        sq += qv; sqq += qv * qv;
        sk += kv; skk += kv * kv;
    }

    __shared__ float red[4][256];
    red[0][tid] = sq; red[1][tid] = sqq; red[2][tid] = sk; red[3][tid] = skk;
    __syncthreads();
    for (int s = 128; s > 0; s >>= 1) {
        if (tid < s) {
            red[0][tid] += red[0][tid + s];
            red[1][tid] += red[1][tid + s];
            red[2][tid] += red[2][tid + s];
            red[3][tid] += red[3][tid + s];
        }
        __syncthreads();
    }

    float invC = 1.0f / (float)C;
    float muq = red[0][0] * invC, mq2 = red[1][0] * invC;
    float muk = red[2][0] * invC, mk2 = red[3][0] * invC;
    float rsq = rsqrtf(mq2 - muq * muq + 1e-5f);
    float rsk = rsqrtf(mk2 - muk * muk + 1e-5f);

    for (int c = tid; c < C; c += 256) {
        int h = c / D, d = c % D;
        size_t oidx = ((size_t)(b * H + h) * N + n) * (size_t)D + d;
        float qv = (row[c]     - muq) * rsq * qg[c] + qbeta[c];
        float kv = (row[C + c] - muk) * rsk * kg[c] + kbeta[c];
        qout[oidx] = (__bf16)qv;
        kout[oidx] = (__bf16)kv;
        vout[oidx] = (__bf16)row[2 * C + c];
    }
}

// ---------------------------------------------------------------------------
// Flash-style attention. Grid: (N/64, B*H), block = 128 (4 waves).
// LDS rows padded to 40 bf16 to break power-of-two bank stride.
// ---------------------------------------------------------------------------
__global__ void __launch_bounds__(128)
attn_kernel(const __bf16* __restrict__ qb, const __bf16* __restrict__ kb,
            const __bf16* __restrict__ vb, __bf16* __restrict__ aout,
            int B, int H, int N, int D) {
    const int C = H * D;
    int bh = blockIdx.y;
    int b = bh / H, h = bh % H;
    int wave = threadIdx.x >> 5;
    int lane = threadIdx.x & 31;
    int col = lane & 15, half = lane >> 4;
    int qRow0 = blockIdx.x * 64 + wave * 16;

    const __bf16* qp = qb + ((size_t)(b * H + h) * N) * (size_t)D;
    const __bf16* kp = kb + ((size_t)(b * H + h) * N) * (size_t)D;
    const __bf16* vp = vb + ((size_t)(b * H + h) * N) * (size_t)D;

    __shared__ __bf16 Vt[64][LDP];      // [d][key-in-block], padded
    __shared__ __bf16 P[4][16][LDP];    // per-wave P tile, padded

    bf16x16 aq0 = load_a_frag(qp + (size_t)qRow0 * D + 0,  D);
    bf16x16 aq1 = load_a_frag(qp + (size_t)qRow0 * D + 32, D);

    f32x8 o[4] = {};
    float mrow[8], lrow[8];
#pragma unroll
    for (int v = 0; v < 8; v++) { mrow[v] = -1e30f; lrow[v] = 0.0f; }
    const float scale = rsqrtf((float)D);

    for (int kb0 = 0; kb0 < N; kb0 += 32) {
        __syncthreads();
        {   // stage V^T: 32 keys x 64 dims
            int kk   = threadIdx.x >> 2;
            int dseg = (threadIdx.x & 3) * 16;
            const __bf16* src = vp + (size_t)(kb0 + kk) * D + dseg;
#pragma unroll
            for (int i = 0; i < 16; i++) Vt[dseg + i][kk] = src[i];
        }
        __syncthreads();

        f32x8 s[2];
#pragma unroll
        for (int ch = 0; ch < 2; ch++) {
            bf16x16 bk0 = load_b_frag(kp + (size_t)(kb0 + ch * 16) * D + 0,  D);
            bf16x16 bk1 = load_b_frag(kp + (size_t)(kb0 + ch * 16) * D + 32, D);
            f32x8 t = {};
            t = wmma_bf16(aq0, bk0, t);
            t = wmma_bf16(aq1, bk1, t);
            s[ch] = t;
        }

#pragma unroll
        for (int v = 0; v < 8; v++) {
            float s0 = s[0][v] * scale;
            float s1 = s[1][v] * scale;
            float mx = fmaxf(s0, s1);
#pragma unroll
            for (int d = 8; d >= 1; d >>= 1) mx = fmaxf(mx, __shfl_xor(mx, d, 32));
            float mnew  = fmaxf(mrow[v], mx);
            float alpha = __expf(mrow[v] - mnew);
            float p0 = __expf(s0 - mnew);
            float p1 = __expf(s1 - mnew);
            float rs = p0 + p1;
#pragma unroll
            for (int d = 8; d >= 1; d >>= 1) rs += __shfl_xor(rs, d, 32);
            lrow[v] = lrow[v] * alpha + rs;
            mrow[v] = mnew;
#pragma unroll
            for (int g = 0; g < 4; g++) o[g][v] *= alpha;
            int r = v + 8 * half;
            P[wave][r][col]      = (__bf16)p0;
            P[wave][r][16 + col] = (__bf16)p1;
        }

        Frag16 pf;
        {
            int r = lane & 15;
            const __bf16* pp = &P[wave][r][half * 8];
            pf.q[0] = *(const uint4*)(pp);
            pf.q[1] = *(const uint4*)(pp + 16);
        }

#pragma unroll
        for (int g = 0; g < 4; g++) {
            Frag16 vf;
            const __bf16* vv = &Vt[g * 16 + col][half * 8];
            vf.q[0] = *(const uint4*)(vv);
            vf.q[1] = *(const uint4*)(vv + 16);
            o[g] = wmma_bf16(pf.v, vf.v, o[g]);
        }
    }

#pragma unroll
    for (int g = 0; g < 4; g++) {
#pragma unroll
        for (int v = 0; v < 8; v++) {
            int r = qRow0 + v + 8 * half;
            int c = h * D + g * 16 + col;
            float inv = 1.0f / lrow[v];
            aout[(size_t)(b * N + r) * C + c] = (__bf16)(o[g][v] * inv);
        }
    }
}

// ---------------------------------------------------------------------------
// Host launcher
// ---------------------------------------------------------------------------
extern "C" void kernel_launch(void* const* d_in, const int* in_sizes, int n_in,
                              void* d_out, int out_size, void* d_ws, size_t ws_size,
                              hipStream_t stream) {
    (void)in_sizes; (void)n_in; (void)out_size; (void)ws_size;

    const float* x       = (const float*)d_in[0];
    const float* w_qkv   = (const float*)d_in[1];
    const float* w_proj  = (const float*)d_in[2];
    const float* b_proj  = (const float*)d_in[3];
    const float* q_gamma = (const float*)d_in[4];
    const float* q_beta  = (const float*)d_in[5];
    const float* k_gamma = (const float*)d_in[6];
    const float* k_beta  = (const float*)d_in[7];

    const int B = 2, N = 2048, C = 1024, H = 16, D = 64;
    const int M = B * N;

    char* p = (char*)d_ws;
    __bf16* xb     = (__bf16*)p; p += (size_t)M * C * 2;
    __bf16* wqkvb  = (__bf16*)p; p += (size_t)3 * C * C * 2;
    __bf16* wprojb = (__bf16*)p; p += (size_t)C * C * 2;
    float*  qkv    = (float*) p; p += (size_t)M * 3 * C * 4;
    __bf16* qbuf   = (__bf16*)p; p += (size_t)M * C * 2;
    __bf16* kbuf   = (__bf16*)p; p += (size_t)M * C * 2;
    __bf16* vbuf   = (__bf16*)p; p += (size_t)M * C * 2;
    __bf16* aob    = (__bf16*)p; p += (size_t)M * C * 2;

    {
        size_t n1 = (size_t)M * C;
        f32_to_bf16_kernel<<<dim3((unsigned)((n1 / 4 + 255) / 256)), dim3(256), 0, stream>>>(x, xb, n1);
        size_t n2 = (size_t)3 * C * C;
        f32_to_bf16_kernel<<<dim3((unsigned)((n2 / 4 + 255) / 256)), dim3(256), 0, stream>>>(w_qkv, wqkvb, n2);
        size_t n3 = (size_t)C * C;
        f32_to_bf16_kernel<<<dim3((unsigned)((n3 / 4 + 255) / 256)), dim3(256), 0, stream>>>(w_proj, wprojb, n3);
    }

    // QKV GEMM: [M,3C] = x[M,C] * w_qkv[3C,C]^T
    gemm_bf16_wmma<<<dim3(M / 128, 3 * C / 128), dim3(256), 0, stream>>>(
        xb, wqkvb, nullptr, qkv, M, 3 * C, C);

    // q/k LayerNorm + scatter to [B,H,N,D] bf16
    qk_ln_kernel<<<dim3(M), dim3(256), 0, stream>>>(
        qkv, q_gamma, q_beta, k_gamma, k_beta, qbuf, kbuf, vbuf, B, N, C, H, D);

    // Flash attention -> bf16 [B,N,C]
    attn_kernel<<<dim3(N / 64, B * H), dim3(128), 0, stream>>>(
        qbuf, kbuf, vbuf, aob, B, H, N, D);

    // Projection GEMM + bias -> f32 d_out
    gemm_bf16_wmma<<<dim3(M / 128, C / 128), dim3(256), 0, stream>>>(
        aob, wprojb, b_proj, (float*)d_out, M, C, C);
}
